// ItemAgg_31267361915503
// MI455X (gfx1250) — matrix operand order repaired and checked by reading, not compiled, not run on hardware
//
#include <hip/hip_runtime.h>
#include <hip/hip_bf16.h>
#include <stdint.h>

// ---------------------------------------------------------------------------
// GraphRec ItemAgg for MI455X (gfx1250, wave32, WMMA 16x16x32 f16)
//
// Pipeline:
//   0) prep_weights   : pack 5 weight matrices into WMMA-B fragment layout (f16)
//   1) edge_mlp       : async-stage weights to LDS (global_load_async_to_lds_b128),
//                       fused 4-layer per-edge MLP, 48 WMMAs per wave-tile,
//                       emits x_ia (f16) and attention logits
//   2) seg_max        : atomicMax (monotone int key) per destination user
//   3) seg_expsum     : exp(logit - max), atomicAdd per-user denominator
//   4) scatter        : h[u] += x_ia * softmax_value   (f32 global atomics)
//   5) out_gemm       : [U,64] @ [64,64] + bias via WMMA
// ---------------------------------------------------------------------------

typedef _Float16 v16h __attribute__((ext_vector_type(16)));
typedef _Float16 v8h  __attribute__((ext_vector_type(8)));
typedef float    v8f  __attribute__((ext_vector_type(8)));

// fragment-packed weight offsets (in halfs) inside workspace / LDS image
constexpr int OFF_GV1  = 0;       // 4n x 4k fragments * 512 halfs = 8192
constexpr int OFF_GV2  = 8192;    // 4n x 2k * 512 = 4096
constexpr int OFF_ATT1 = 12288;   // 8192
constexpr int OFF_ATT2 = 20480;   // 4096
constexpr int OFF_WW   = 24576;   // 4096
constexpr int WFRAG_HALFS = 28672;
constexpr int WFRAG_BYTES = WFRAG_HALFS * 2;          // 57344 = 14 * 256 * 16
constexpr int EDGE_LDS_BYTES = WFRAG_BYTES + 8 * 16 * 64 * 2;   // + 8 wave tiles

// ---- WMMA fragment helpers ------------------------------------------------
// A (16x32 f16): lane m = l%16; half j holds K = j + (j&8) + 8*hi  (ISA 7.12.2)
// B (32x16 f16): lane n = l%16; half j holds K = j + (j&8) + 8*hi  (same as A)
// C (16x16 f32): vgpr g holds (M = g + 8*hi, N = l%16)

static __device__ __forceinline__ v16h make_a_frag_f32row(const float* __restrict__ row, int hi)
{
    // row points at the 32 consecutive source columns of this K-chunk
    v16h a;
#pragma unroll
    for (int j = 0; j < 16; j += 2) {
        const int k = j + (j & 8) + (hi << 3);          // even -> 8B aligned pair
        const float2 p = *(const float2*)(row + k);
        a[j]     = (_Float16)p.x;
        a[j + 1] = (_Float16)p.y;
    }
    return a;
}

static __device__ __forceinline__ v16h make_a_frag_lds(const _Float16* __restrict__ tile_row, int hi)
{
    // tile_row = &tile[m*64 + chunk*32]; K runs are contiguous 8-half groups
    v16h a;
    const v8h lo = *(const v8h*)(tile_row + 8 * hi);        // K = 8*hi .. 8*hi+7
    const v8h up = *(const v8h*)(tile_row + 16 + 8 * hi);   // K = 16+8*hi .. 23+8*hi
#pragma unroll
    for (int j = 0; j < 8; ++j) { a[j] = lo[j]; a[j + 8] = up[j]; }
    return a;
}

static __device__ __forceinline__ v16h load_b_frag(const _Float16* __restrict__ wbase, int fragIdx, int lane)
{
    return *(const v16h*)(wbase + (size_t)fragIdx * 512 + lane * 16);   // 32B per lane
}

// ---- kernel 0: pack weights into B-fragment layout ------------------------
__global__ void __launch_bounds__(256)
prep_weights(const float* __restrict__ gv_w1, const float* __restrict__ gv_w2,
             const float* __restrict__ att_w1, const float* __restrict__ att_w2,
             const float* __restrict__ w_w, _Float16* __restrict__ wfrag)
{
    int t = blockIdx.x * blockDim.x + threadIdx.x;
    if (t >= WFRAG_HALFS) return;
    const float* src; int K; int o = t;
    if      (o < OFF_GV2)  { src = gv_w1;  K = 128; o -= OFF_GV1;  }
    else if (o < OFF_ATT1) { src = gv_w2;  K = 64;  o -= OFF_GV2;  }
    else if (o < OFF_ATT2) { src = att_w1; K = 128; o -= OFF_ATT1; }
    else if (o < OFF_WW)   { src = att_w2; K = 64;  o -= OFF_ATT2; }
    else                   { src = w_w;    K = 64;  o -= OFF_WW;   }
    const int j    = o & 15;
    const int lane = (o >> 4) & 31;
    const int f    = o >> 9;
    const int nC   = K >> 5;
    const int kc   = f % nC;
    const int n    = f / nC;
    const int krow = kc * 32 + j + (j & 8) + ((lane >= 16) ? 8 : 0);
    const int col  = n * 16 + (lane & 15);
    wfrag[t] = (_Float16)src[krow * 64 + col];
}

// ---- kernel 1: fused per-edge MLP -----------------------------------------
__global__ void __launch_bounds__(256)
edge_mlp_kernel(const float* __restrict__ user_feat,
                const float* __restrict__ item_feat,
                const float* __restrict__ rating_feat,
                const int* __restrict__ row_idxs,
                const int* __restrict__ col_idxs,
                const int* __restrict__ rating,
                const float* __restrict__ gv_b1, const float* __restrict__ gv_b2,
                const float* __restrict__ att_b1, const float* __restrict__ att_b2,
                const float* __restrict__ att_w3, const float* __restrict__ att_b3,
                const _Float16* __restrict__ wfrag_g,
                _Float16* __restrict__ xia_out,   // [E,64] f16
                float* __restrict__ logits,       // [E]
                int E)
{
    extern __shared__ char smem[];
    _Float16* wlds  = (_Float16*)smem;                       // 56KB weight image
    _Float16* tiles = (_Float16*)(smem + WFRAG_BYTES);       // 8 x (16x64) f16 tiles

    // ---- stage fragment-packed weights to LDS with async copies ----------
    {
        const int t = threadIdx.x;
#pragma unroll
        for (int it = 0; it < WFRAG_BYTES / (256 * 16); ++it) {   // 14 iterations
            const int byteOff = (it * 256 + t) * 16;
            const unsigned lds_off =
                (unsigned)(uintptr_t)((char*)wlds + byteOff);     // LDS aperture: offset in addr[31:0]
            const unsigned long long gaddr =
                (unsigned long long)((const char*)wfrag_g + byteOff);
            asm volatile("global_load_async_to_lds_b128 %0, %1, off"
                         :: "v"(lds_off), "v"(gaddr) : "memory");
        }
        asm volatile("s_wait_asynccnt 0x0" ::: "memory");
        __syncthreads();
    }

    const int wave = threadIdx.x >> 5;
    const int l    = threadIdx.x & 31;
    const int l16  = l & 15;
    const int hi   = l >> 4;
    const int tileIdx = blockIdx.x * 8 + wave;     // uniform per wave
    const int ebase   = tileIdx * 16;
    if (ebase >= E) return;                        // whole-wave exit: EXEC stays all-1s below
    _Float16* tile = tiles + wave * (16 * 64);

    // this lane's A-matrix row (edge)
    int e = ebase + l16; if (e >= E) e = E - 1;
    const float* irow = item_feat   + (size_t)row_idxs[e] * 64;
    const float* urow = user_feat   + (size_t)col_idxs[e] * 64;
    const float* rrow = rating_feat + (size_t)rating[e]   * 64;

    v8f acc[4];

    // ======== gv layer 1: [i_emb | r_emb] (16x128) @ gv_w1 -> relu ========
    v16h a1[4];
    a1[0] = make_a_frag_f32row(irow,      hi);
    a1[1] = make_a_frag_f32row(irow + 32, hi);
    a1[2] = make_a_frag_f32row(rrow,      hi);
    a1[3] = make_a_frag_f32row(rrow + 32, hi);
#pragma unroll
    for (int nt = 0; nt < 4; ++nt) {
        const float b = gv_b1[nt * 16 + l16];
        v8f c;
#pragma unroll
        for (int g = 0; g < 8; ++g) c[g] = b;
#pragma unroll
        for (int kc = 0; kc < 4; ++kc) {
            const v16h bf = load_b_frag(wlds + OFF_GV1, nt * 4 + kc, l);
            c = __builtin_amdgcn_wmma_f32_16x16x32_f16(false, a1[kc], false, bf, (short)0, c, false, false);
        }
#pragma unroll
        for (int g = 0; g < 8; ++g) c[g] = c[g] > 0.f ? c[g] : 0.f;
        acc[nt] = c;
    }
    // transpose C->A via wave-private LDS tile (row-major 16x64 f16)
#pragma unroll
    for (int nt = 0; nt < 4; ++nt)
#pragma unroll
        for (int g = 0; g < 8; ++g)
            tile[(g + 8 * hi) * 64 + nt * 16 + l16] = (_Float16)acc[nt][g];

    // ======== gv layer 2: h1 (16x64) @ gv_w2 -> relu = x_ia ========
    v16h a2[2];
    a2[0] = make_a_frag_lds(tile + l16 * 64,      hi);
    a2[1] = make_a_frag_lds(tile + l16 * 64 + 32, hi);
#pragma unroll
    for (int nt = 0; nt < 4; ++nt) {
        const float b = gv_b2[nt * 16 + l16];
        v8f c;
#pragma unroll
        for (int g = 0; g < 8; ++g) c[g] = b;
#pragma unroll
        for (int kc = 0; kc < 2; ++kc) {
            const v16h bf = load_b_frag(wlds + OFF_GV2, nt * 2 + kc, l);
            c = __builtin_amdgcn_wmma_f32_16x16x32_f16(false, a2[kc], false, bf, (short)0, c, false, false);
        }
#pragma unroll
        for (int g = 0; g < 8; ++g) c[g] = c[g] > 0.f ? c[g] : 0.f;
        acc[nt] = c;
    }
    // x_ia: write to LDS (for attention A-build) and to global (f16, for scatter pass)
#pragma unroll
    for (int nt = 0; nt < 4; ++nt)
#pragma unroll
        for (int g = 0; g < 8; ++g) {
            const _Float16 hv = (_Float16)acc[nt][g];
            tile[(g + 8 * hi) * 64 + nt * 16 + l16] = hv;
            const int er = ebase + g + 8 * hi;
            if (er < E) xia_out[(size_t)er * 64 + nt * 16 + l16] = hv;
        }

    // ======== attention layer 1: [x_ia | u_emb] (16x128) @ att_w1 -> relu ========
    v16h a3[4];
    a3[0] = make_a_frag_lds(tile + l16 * 64,      hi);
    a3[1] = make_a_frag_lds(tile + l16 * 64 + 32, hi);
    a3[2] = make_a_frag_f32row(urow,      hi);
    a3[3] = make_a_frag_f32row(urow + 32, hi);
#pragma unroll
    for (int nt = 0; nt < 4; ++nt) {
        const float b = att_b1[nt * 16 + l16];
        v8f c;
#pragma unroll
        for (int g = 0; g < 8; ++g) c[g] = b;
#pragma unroll
        for (int kc = 0; kc < 4; ++kc) {
            const v16h bf = load_b_frag(wlds + OFF_ATT1, nt * 4 + kc, l);
            c = __builtin_amdgcn_wmma_f32_16x16x32_f16(false, a3[kc], false, bf, (short)0, c, false, false);
        }
#pragma unroll
        for (int g = 0; g < 8; ++g) c[g] = c[g] > 0.f ? c[g] : 0.f;
        acc[nt] = c;
    }
#pragma unroll
    for (int nt = 0; nt < 4; ++nt)
#pragma unroll
        for (int g = 0; g < 8; ++g)
            tile[(g + 8 * hi) * 64 + nt * 16 + l16] = (_Float16)acc[nt][g];

    // ======== attention layer 2: (16x64) @ att_w2 -> relu ========
    v16h a4[2];
    a4[0] = make_a_frag_lds(tile + l16 * 64,      hi);
    a4[1] = make_a_frag_lds(tile + l16 * 64 + 32, hi);
#pragma unroll
    for (int nt = 0; nt < 4; ++nt) {
        const float b = att_b2[nt * 16 + l16];
        v8f c;
#pragma unroll
        for (int g = 0; g < 8; ++g) c[g] = b;
#pragma unroll
        for (int kc = 0; kc < 2; ++kc) {
            const v16h bf = load_b_frag(wlds + OFF_ATT2, nt * 2 + kc, l);
            c = __builtin_amdgcn_wmma_f32_16x16x32_f16(false, a4[kc], false, bf, (short)0, c, false, false);
        }
#pragma unroll
        for (int g = 0; g < 8; ++g) c[g] = c[g] > 0.f ? c[g] : 0.f;
        acc[nt] = c;
    }

    // ======== attention head: a @ att_w3 + b3 (lane dot + 16-lane butterfly) ========
    float p[8];
#pragma unroll
    for (int g = 0; g < 8; ++g) p[g] = 0.f;
#pragma unroll
    for (int nt = 0; nt < 4; ++nt) {
        const float w3 = att_w3[nt * 16 + l16];
#pragma unroll
        for (int g = 0; g < 8; ++g) p[g] += acc[nt][g] * w3;
    }
#pragma unroll
    for (int off = 8; off > 0; off >>= 1)
#pragma unroll
        for (int g = 0; g < 8; ++g) p[g] += __shfl_xor(p[g], off, 32);

    const float b3 = att_b3[0];
#pragma unroll
    for (int g = 0; g < 8; ++g) {
        if (l16 == g) {
            const int er = ebase + g + 8 * hi;
            if (er < E) logits[er] = p[g] + b3;
        }
    }
}

// ---- kernel 2: segment max via monotone-int-key atomicMax -----------------
__global__ void __launch_bounds__(256)
seg_max_kernel(const float* __restrict__ logits, const int* __restrict__ col,
               unsigned* __restrict__ maxkey, int E)
{
    const int e = blockIdx.x * blockDim.x + threadIdx.x;
    if (e >= E) return;
    const unsigned b = __float_as_uint(logits[e]);
    const unsigned key = (b & 0x80000000u) ? ~b : (b | 0x80000000u);
    atomicMax(maxkey + col[e], key);
}

// ---- kernel 3: exp + segment denominator ----------------------------------
__global__ void __launch_bounds__(256)
seg_expsum_kernel(const float* __restrict__ logits, const int* __restrict__ col,
                  const unsigned* __restrict__ maxkey,
                  float* __restrict__ ebuf, float* __restrict__ sums, int E)
{
    const int e = blockIdx.x * blockDim.x + threadIdx.x;
    if (e >= E) return;
    const int u = col[e];
    const unsigned k = maxkey[u];
    const unsigned b = (k & 0x80000000u) ? (k & 0x7fffffffu) : ~k;
    const float m = __uint_as_float(b);
    const float ev = __expf(logits[e] - m);
    ebuf[e] = ev;
    atomicAdd(sums + u, ev);
}

// ---- kernel 4: weighted scatter-sum to users ------------------------------
__global__ void __launch_bounds__(256)
scatter_kernel(const _Float16* __restrict__ xia, const float* __restrict__ ebuf,
               const float* __restrict__ sums, const int* __restrict__ col,
               float* __restrict__ hbuf, long long EN)
{
    const long long g = blockIdx.x * (long long)blockDim.x + threadIdx.x;
    if (g >= EN) return;
    const int e = (int)(g >> 6);
    const int d = (int)(g & 63);
    const int u = col[e];
    const float v = ebuf[e] / sums[u];
    atomicAdd(hbuf + (size_t)u * 64 + d, (float)xia[(size_t)e * 64 + d] * v);
}

// ---- kernel 5: output projection [U,64] @ [64,64] + bias via WMMA ----------
__global__ void __launch_bounds__(256)
out_gemm_kernel(const float* __restrict__ hbuf, const _Float16* __restrict__ wfrag,
                const float* __restrict__ w_b, float* __restrict__ out, int U)
{
    const int wave = threadIdx.x >> 5;
    const int l    = threadIdx.x & 31;
    const int l16  = l & 15;
    const int hi   = l >> 4;
    const int tileIdx = blockIdx.x * 8 + wave;
    const int ub = tileIdx * 16;
    if (ub >= U) return;
    int ur = ub + l16; if (ur >= U) ur = U - 1;
    const float* hrow = hbuf + (size_t)ur * 64;
    v16h a[2];
    a[0] = make_a_frag_f32row(hrow,      hi);
    a[1] = make_a_frag_f32row(hrow + 32, hi);
#pragma unroll
    for (int nt = 0; nt < 4; ++nt) {
        const float b = w_b[nt * 16 + l16];
        v8f c;
#pragma unroll
        for (int g = 0; g < 8; ++g) c[g] = b;
#pragma unroll
        for (int kc = 0; kc < 2; ++kc) {
            const v16h bf = load_b_frag(wfrag + OFF_WW, nt * 2 + kc, l);
            c = __builtin_amdgcn_wmma_f32_16x16x32_f16(false, a[kc], false, bf, (short)0, c, false, false);
        }
#pragma unroll
        for (int g = 0; g < 8; ++g) {
            const int row = ub + g + 8 * hi;
            if (row < U) out[(size_t)row * 64 + nt * 16 + l16] = c[g];
        }
    }
}

// ---------------------------------------------------------------------------
extern "C" void kernel_launch(void* const* d_in, const int* in_sizes, int n_in,
                              void* d_out, int out_size, void* d_ws, size_t ws_size,
                              hipStream_t stream)
{
    const float* user_feat   = (const float*)d_in[0];
    const float* item_feat   = (const float*)d_in[1];
    const float* rating_feat = (const float*)d_in[2];
    const int*   row_idxs    = (const int*)d_in[3];
    const int*   col_idxs    = (const int*)d_in[4];
    const int*   rating      = (const int*)d_in[5];
    const float* gv_w1  = (const float*)d_in[6];
    const float* gv_b1  = (const float*)d_in[7];
    const float* gv_w2  = (const float*)d_in[8];
    const float* gv_b2  = (const float*)d_in[9];
    const float* att_w1 = (const float*)d_in[10];
    const float* att_b1 = (const float*)d_in[11];
    const float* att_w2 = (const float*)d_in[12];
    const float* att_b2 = (const float*)d_in[13];
    const float* att_w3 = (const float*)d_in[14];
    const float* att_b3 = (const float*)d_in[15];
    const float* w_w    = (const float*)d_in[16];
    const float* w_b    = (const float*)d_in[17];

    const int U = in_sizes[0] / 64;
    const int E = in_sizes[3];

    char* ws = (char*)d_ws;
    size_t off = 0;
    auto carve = [&](size_t bytes) {
        char* p = ws + off;
        off = (off + bytes + 255) & ~(size_t)255;
        return p;
    };
    _Float16* wfrag  = (_Float16*)carve((size_t)WFRAG_HALFS * sizeof(_Float16));
    _Float16* xia    = (_Float16*)carve((size_t)E * 64 * sizeof(_Float16));
    float*    logits = (float*)   carve((size_t)E * sizeof(float));
    float*    ebuf   = (float*)   carve((size_t)E * sizeof(float));
    unsigned* maxkey = (unsigned*)carve((size_t)U * sizeof(unsigned));
    float*    sums   = (float*)   carve((size_t)U * sizeof(float));
    float*    hbuf   = (float*)   carve((size_t)U * 64 * sizeof(float));
    (void)ws_size; (void)n_in; (void)out_size;

    // deterministic per-call init of accumulators (harness does not re-poison)
    hipMemsetAsync(maxkey, 0, (size_t)U * sizeof(unsigned), stream);   // key(-inf) encoding safe
    hipMemsetAsync(sums,   0, (size_t)U * sizeof(float), stream);
    hipMemsetAsync(hbuf,   0, (size_t)U * 64 * sizeof(float), stream);

    prep_weights<<<(WFRAG_HALFS + 255) / 256, 256, 0, stream>>>(gv_w1, gv_w2, att_w1, att_w2, w_w, wfrag);

    const int etiles = (E + 15) / 16;
    edge_mlp_kernel<<<(etiles + 7) / 8, 256, EDGE_LDS_BYTES, stream>>>(
        user_feat, item_feat, rating_feat, row_idxs, col_idxs, rating,
        gv_b1, gv_b2, att_b1, att_b2, att_w3, att_b3, wfrag, xia, logits, E);

    seg_max_kernel<<<(E + 255) / 256, 256, 0, stream>>>(logits, col_idxs, maxkey, E);
    seg_expsum_kernel<<<(E + 255) / 256, 256, 0, stream>>>(logits, col_idxs, maxkey, ebuf, sums, E);

    const long long EN = (long long)E * 64;
    scatter_kernel<<<(unsigned)((EN + 255) / 256), 256, 0, stream>>>(xia, ebuf, sums, col_idxs, hbuf, EN);

    const int utiles = (U + 15) / 16;
    out_gemm_kernel<<<(utiles + 7) / 8, 256, 0, stream>>>(hbuf, wfrag, w_b, (float*)d_out, U);
}
